// SimpleLSTMHedger_81106162418317
// MI455X (gfx1250) — matrix operand order repaired and checked by reading, not compiled, not run on hardware
//
#include <hip/hip_runtime.h>

typedef __attribute__((ext_vector_type(16))) _Float16 v16h;
typedef __attribute__((ext_vector_type(8)))  float    v8f;

namespace cfg {
constexpr int DIN  = 5;     // x features per step
constexpr int K0IN = 6;     // layer0 input features = DIN + prev_delta
constexpr int H    = 50;    // hidden size
constexpr int NG   = 200;   // 4*H fused gates
constexpr int NPAD = 208;   // gates padded to 13 WMMA N-tiles
constexpr int NT   = 13;
constexpr int K0P  = 64;    // (6 + 50) padded for 2 k-steps of 32
constexpr int K1P  = 128;   // (50 + 50) padded for 4 k-steps of 32
constexpr int T    = 512;
constexpr int WAVES = 4;    // waves per workgroup
constexpr int RPW  = 16;    // batch rows per wave (WMMA M)
constexpr int RPB  = WAVES * RPW;   // 64 batch rows per workgroup
constexpr int NTHR = WAVES * 32;    // 128 threads

// --- LDS layout (bytes) ---
constexpr unsigned OFF_W0 = 0;                         // f16 [NPAD][K0P]
constexpr unsigned OFF_W1 = OFF_W0 + NPAD * K0P * 2;   // f16 [NPAD][K1P]
constexpr unsigned OFF_B0 = OFF_W1 + NPAD * K1P * 2;   // f32 [NPAD]
constexpr unsigned OFF_B1 = OFF_B0 + NPAD * 4;         // f32 [NPAD]
constexpr unsigned OFF_WO = OFF_B1 + NPAD * 4;         // f32 [H] + b_out
constexpr unsigned OFF_PW = ((OFF_WO + 64 * 4 + 255u) / 256u) * 256u;

// per-wave block (bytes)
constexpr unsigned PW_A0 = 0;                          // f16 [RPW][K0P] layer0 input
constexpr unsigned PW_A1 = PW_A0 + RPW * K0P * 2;      // f16 [RPW][K1P] layer1 input
constexpr unsigned PW_Z  = PW_A1 + RPW * K1P * 2;      // f32 [RPW][NPAD] gate pre-acts
constexpr unsigned PW_C0 = PW_Z  + RPW * NPAD * 4;     // f32 [RPW][H]
constexpr unsigned PW_C1 = PW_C0 + RPW * H * 4;        // f32 [RPW][H]
constexpr unsigned PW_H1 = PW_C1 + RPW * H * 4;        // f32 [RPW][H]
constexpr unsigned PW_SZ = ((PW_H1 + RPW * H * 4 + 255u) / 256u) * 256u;

constexpr unsigned SMEM_BYTES = OFF_PW + WAVES * PW_SZ;
static_assert(SMEM_BYTES <= 320u * 1024u, "exceeds WGP LDS");
}

__device__ __forceinline__ float sigf(float x) { return 1.0f / (1.0f + __expf(-x)); }

// Wave-local LDS producer->consumer ordering. LDS instructions from a single
// wave are processed in order (DScnt is in-order per ISA), so cross-lane
// visibility within the wave needs only (a) the outstanding DS ops drained /
// made visible (fence emits s_wait_dscnt) and (b) a compiler scheduling
// barrier so LDS ops are not reordered across the sync point. No s_barrier
// needed -> the 4 waves of the workgroup run fully decoupled.
__device__ __forceinline__ void wave_sync() {
    __builtin_amdgcn_fence(__ATOMIC_RELEASE, "workgroup");
    __builtin_amdgcn_wave_barrier();
    __builtin_amdgcn_fence(__ATOMIC_ACQUIRE, "workgroup");
}

// Load a v16h fragment from two 16-byte LDS chunks.
__device__ __forceinline__ v16h lds_frag(const _Float16* p0, const _Float16* p1) {
    union { uint4 q[2]; v16h v; } u;
    u.q[0] = *(const uint4*)p0;
    u.q[1] = *(const uint4*)p1;
    return u.v;
}

// A-matrix fragment (16xK row-major in LDS). ISA layout: lanes 0-15 = rows,
// lane holds K [g*8, g*8+8) then [16+g*8, 16+g*8+8) within each 32-wide k-tile.
__device__ __forceinline__ v16h ldsA(const _Float16* row, int kt, int g) {
    const _Float16* p = row + kt * 32 + g * 8;
    return lds_frag(p, p + 16);
}

// B-matrix fragment (weights stored gate-major [n][k]). Lane = column n,
// holds K [g*8, g*8+8) then [16+g*8, 16+g*8+8) within the k-tile.
__device__ __forceinline__ v16h ldsB(const _Float16* row, int kt, int g) {
    const _Float16* p = row + kt * 32 + g * 8;
    return lds_frag(p, p + 16);
}

__global__ __launch_bounds__(cfg::NTHR)
void lstm_hedger_kernel(const float* __restrict__ x,
                        const float* __restrict__ w_ih0, const float* __restrict__ w_hh0,
                        const float* __restrict__ b_ih0, const float* __restrict__ b_hh0,
                        const float* __restrict__ w_ih1, const float* __restrict__ w_hh1,
                        const float* __restrict__ b_ih1, const float* __restrict__ b_hh1,
                        const float* __restrict__ w_out, const float* __restrict__ b_out,
                        float* __restrict__ out) {
    extern __shared__ char smem[];
    const int tid  = threadIdx.x;
    const int wave = tid >> 5;
    const int lane = tid & 31;
    const int g    = lane >> 4;   // half-wave group (WMMA layout)
    const int l16  = lane & 15;

    _Float16* W0 = (_Float16*)(smem + cfg::OFF_W0);
    _Float16* W1 = (_Float16*)(smem + cfg::OFF_W1);
    float* Bias0 = (float*)(smem + cfg::OFF_B0);
    float* Bias1 = (float*)(smem + cfg::OFF_B1);
    float* WO    = (float*)(smem + cfg::OFF_WO);   // [0..H-1]=w_out, [H]=b_out

    // ---- one-time: pack weights f32 -> f16 into LDS, fused & padded ----
    for (int idx = tid; idx < cfg::NPAD * cfg::K0P; idx += cfg::NTHR) {
        int n = idx >> 6, k = idx & 63;
        float v = 0.0f;
        if (n < cfg::NG) {
            if (k < cfg::K0IN)                 v = w_ih0[n * cfg::K0IN + k];
            else if (k < cfg::K0IN + cfg::H)   v = w_hh0[n * cfg::H + (k - cfg::K0IN)];
        }
        W0[idx] = (_Float16)v;
    }
    for (int idx = tid; idx < cfg::NPAD * cfg::K1P; idx += cfg::NTHR) {
        int n = idx >> 7, k = idx & 127;
        float v = 0.0f;
        if (n < cfg::NG) {
            if (k < cfg::H)            v = w_ih1[n * cfg::H + k];
            else if (k < 2 * cfg::H)   v = w_hh1[n * cfg::H + (k - cfg::H)];
        }
        W1[idx] = (_Float16)v;
    }
    for (int n = tid; n < cfg::NPAD; n += cfg::NTHR) {
        Bias0[n] = (n < cfg::NG) ? (b_ih0[n] + b_hh0[n]) : 0.0f;
        Bias1[n] = (n < cfg::NG) ? (b_ih1[n] + b_hh1[n]) : 0.0f;
    }
    if (tid < cfg::H) WO[tid] = w_out[tid];
    if (tid == cfg::H) WO[cfg::H] = b_out[0];
    // zero all per-wave state (h=c=delta=0, pads stay 0 forever)
    for (unsigned i = (unsigned)tid * 4u; i < cfg::WAVES * cfg::PW_SZ; i += cfg::NTHR * 4u)
        *(float*)(smem + cfg::OFF_PW + i) = 0.0f;
    __syncthreads();   // weights become read-only; waves decouple from here on

    char* pw = smem + cfg::OFF_PW + (unsigned)wave * cfg::PW_SZ;
    _Float16* A0 = (_Float16*)(pw + cfg::PW_A0);
    _Float16* A1 = (_Float16*)(pw + cfg::PW_A1);
    float*    Z  = (float*)(pw + cfg::PW_Z);
    float*    C0 = (float*)(pw + cfg::PW_C0);
    float*    C1 = (float*)(pw + cfg::PW_C1);
    float*    H1 = (float*)(pw + cfg::PW_H1);

    const int bbase = blockIdx.x * cfg::RPB + wave * cfg::RPW;
    const v8f vzero = {0.f, 0.f, 0.f, 0.f, 0.f, 0.f, 0.f, 0.f};

    for (int t = 0; t < cfg::T; ++t) {
        // 1) stage x_t into A0[:, 0..4]; A0[:,5] = prev_delta (written last step)
        if (lane < cfg::RPW) {
            const float* xp = x + ((size_t)(bbase + lane) * cfg::T + t) * cfg::DIN;
            #pragma unroll
            for (int i = 0; i < cfg::DIN; ++i)
                A0[lane * cfg::K0P + i] = (_Float16)xp[i];
            if (t + 1 < cfg::T) __builtin_prefetch(xp + cfg::DIN, 0, 3);
        }
        wave_sync();

        // 2) layer0 gates: Z[16 x 208] = A0[16 x 64] @ W0^T (+bias at store)
        for (int nt = 0; nt < cfg::NT; ++nt) {
            v8f acc = vzero;
            #pragma unroll
            for (int kt = 0; kt < 2; ++kt) {
                v16h a = ldsA(A0 + l16 * cfg::K0P, kt, g);
                v16h w = ldsB(W0 + (nt * 16 + l16) * cfg::K0P, kt, g);
                acc = __builtin_amdgcn_wmma_f32_16x16x32_f16(
                          false, a, false, w, (short)0, acc, false, false);
            }
            const int n = nt * 16 + l16;
            const float bv = Bias0[n];
            #pragma unroll
            for (int r = 0; r < 8; ++r)           // D layout: vgpr r -> M=r+8g
                Z[(r + 8 * g) * cfg::NPAD + n] = acc[r] + bv;
        }
        wave_sync();

        // 3) layer0 cell update; h0 feeds A1[:,0..49] and next-step A0[:,6..55]
        for (int idx = lane; idx < cfg::RPW * cfg::H; idx += 32) {
            const int m = idx / cfg::H, j = idx - m * cfg::H;
            const float zi = Z[m * cfg::NPAD + j];
            const float zf = Z[m * cfg::NPAD + cfg::H + j];
            const float zg = Z[m * cfg::NPAD + 2 * cfg::H + j];
            const float zo = Z[m * cfg::NPAD + 3 * cfg::H + j];
            const float c  = sigf(zf) * C0[idx] + sigf(zi) * tanhf(zg);
            C0[idx] = c;
            const _Float16 hh = (_Float16)(sigf(zo) * tanhf(c));
            A0[m * cfg::K0P + cfg::K0IN + j] = hh;
            A1[m * cfg::K1P + j] = hh;
        }
        wave_sync();

        // 4) layer1 gates: Z = A1[16 x 128] @ W1^T (+bias at store)
        for (int nt = 0; nt < cfg::NT; ++nt) {
            v8f acc = vzero;
            #pragma unroll
            for (int kt = 0; kt < 4; ++kt) {
                v16h a = ldsA(A1 + l16 * cfg::K1P, kt, g);
                v16h w = ldsB(W1 + (nt * 16 + l16) * cfg::K1P, kt, g);
                acc = __builtin_amdgcn_wmma_f32_16x16x32_f16(
                          false, a, false, w, (short)0, acc, false, false);
            }
            const int n = nt * 16 + l16;
            const float bv = Bias1[n];
            #pragma unroll
            for (int r = 0; r < 8; ++r)
                Z[(r + 8 * g) * cfg::NPAD + n] = acc[r] + bv;
        }
        wave_sync();

        // 5) layer1 cell update; h1 -> H1 (f32 for output dot) and A1[:,50..99]
        for (int idx = lane; idx < cfg::RPW * cfg::H; idx += 32) {
            const int m = idx / cfg::H, j = idx - m * cfg::H;
            const float zi = Z[m * cfg::NPAD + j];
            const float zf = Z[m * cfg::NPAD + cfg::H + j];
            const float zg = Z[m * cfg::NPAD + 2 * cfg::H + j];
            const float zo = Z[m * cfg::NPAD + 3 * cfg::H + j];
            const float c  = sigf(zf) * C1[idx] + sigf(zi) * tanhf(zg);
            C1[idx] = c;
            const float h = sigf(zo) * tanhf(c);
            H1[idx] = h;
            A1[m * cfg::K1P + cfg::H + j] = (_Float16)h;
        }
        wave_sync();

        // 6) delta = 1.5*tanh(h1 . w_out + b_out); emit + feed back
        if (lane < cfg::RPW) {
            float acc = WO[cfg::H];
            const float* hr = H1 + lane * cfg::H;
            #pragma unroll 5
            for (int j = 0; j < cfg::H; ++j) acc += hr[j] * WO[j];
            const float delta = 1.5f * tanhf(acc);
            out[(size_t)(bbase + lane) * cfg::T + t] = delta;
            A0[lane * cfg::K0P + cfg::DIN] = (_Float16)delta;
        }
        wave_sync();
    }
}

extern "C" void kernel_launch(void* const* d_in, const int* in_sizes, int n_in,
                              void* d_out, int out_size, void* d_ws, size_t ws_size,
                              hipStream_t stream) {
    const float* x     = (const float*)d_in[0];
    const float* w_ih0 = (const float*)d_in[1];
    const float* w_hh0 = (const float*)d_in[2];
    const float* b_ih0 = (const float*)d_in[3];
    const float* b_hh0 = (const float*)d_in[4];
    const float* w_ih1 = (const float*)d_in[5];
    const float* w_hh1 = (const float*)d_in[6];
    const float* b_ih1 = (const float*)d_in[7];
    const float* b_hh1 = (const float*)d_in[8];
    const float* w_out = (const float*)d_in[9];
    const float* b_out = (const float*)d_in[10];
    float* out = (float*)d_out;

    const int B = in_sizes[0] / (cfg::T * cfg::DIN);   // 16384
    dim3 grid(B / cfg::RPB);                           // 256 workgroups
    dim3 block(cfg::NTHR);                             // 4 waves (wave32)
    lstm_hedger_kernel<<<grid, block, cfg::SMEM_BYTES, stream>>>(
        x, w_ih0, w_hh0, b_ih0, b_hh0, w_ih1, w_hh1, b_ih1, b_hh1,
        w_out, b_out, out);
}
